// HG_LGCN_1279900254485
// MI455X (gfx1250) — matrix operand order repaired and checked by reading, not compiled, not run on hardware
//
#include <hip/hip_runtime.h>

typedef __attribute__((ext_vector_type(16))) _Float16 v16h;
typedef __attribute__((ext_vector_type(8)))  _Float16 v8h;
typedef __attribute__((ext_vector_type(8)))  float    v8f;

#define NNODES 50000
#define NEDGES 800000
#define INDIM  16
#define HID    512
#define OUTD   64
#define CONCAT 496
#define CPAD   512   // K padded to multiple of 32

// ---------- workspace layout (bytes) ----------
#define HC_OFF   ((size_t)0)                          // f32 [N,512] concat feats; later reused as h1
#define HH_OFF   (HC_OFF  + (size_t)NNODES*CPAD*4)    // f16 [N,512]; later reused as BN-applied f16
#define Z_OFF    (HH_OFF  + (size_t)NNODES*CPAD*2)    // f32 [N,64]
#define W1P_OFF  (Z_OFF   + (size_t)NNODES*OUTD*4)    // f16 packed W1  (512*512)
#define W2P_OFF  (W1P_OFF + (size_t)HID*CPAD*2)       // f16 packed W2  (64*512)
#define SUM_OFF  (W2P_OFF + (size_t)OUTD*HID*2)       // f32 [512]
#define SQ_OFF   (SUM_OFF + 2048)
#define SC_OFF   (SQ_OFF  + 2048)
#define SH_OFF   (SC_OFF  + 2048)

// ---------- copy feature into columns [0,16) of padded concat buffer ----------
__global__ void k_feat(const float* __restrict__ f, float* __restrict__ Hc) {
  int gid = blockIdx.x * blockDim.x + threadIdx.x;
  if (gid >= NNODES * INDIM) return;
  int n = gid >> 4, d = gid & 15;
  Hc[(size_t)n * CPAD + d] = f[gid];
}

// ---------- one propagation round: weighted gather/scatter-add by edge type ----------
__global__ void k_scatter(const int* __restrict__ src, const int* __restrict__ dst,
                          const float* __restrict__ ew, const int* __restrict__ et,
                          float* __restrict__ Hc, int Din, int baseIn, int baseOut, int lg) {
  int gid = blockIdx.x * blockDim.x + threadIdx.x;
  int chunks = Din >> 2;
  if (gid >= NEDGES * chunks) return;
  int e = gid >> lg;            // lg = log2(chunks)
  int c = gid & (chunks - 1);
  int s = src[e], d = dst[e];
  float w = ew[e];
  int t = (et[e] != 0) ? 1 : 0;
  const float4 x = *(const float4*)(Hc + (size_t)s * CPAD + baseIn + c * 4);
  float* o = Hc + (size_t)d * CPAD + baseOut + t * Din + c * 4;
  unsafeAtomicAdd(o + 0, w * x.x);
  unsafeAtomicAdd(o + 1, w * x.y);
  unsafeAtomicAdd(o + 2, w * x.z);
  unsafeAtomicAdd(o + 3, w * x.w);
}

// ---------- pack a [J, Creal] f32 weight into WMMA B-fragment layout (f16) ----------
// B-layout: lane = N%16 + 16*((K>>3)&1), element = (K&7) + 8*(K>>4) (K within 32-step).
// P[((jt*nks + ks)*32 + lane)*16 + e]
__global__ void k_packB(const float* __restrict__ W, _Float16* __restrict__ P,
                        int J, int Creal, int Cpad) {
  int gid = blockIdx.x * blockDim.x + threadIdx.x;
  if (gid >= J * Cpad) return;
  int j = gid / Cpad, c = gid % Cpad;
  float v = (c < Creal) ? W[(size_t)j * Creal + c] : 0.0f;
  int jt = j >> 4, ks = c >> 5, kk = c & 31;
  int lane = (((kk >> 3) & 1) << 4) + (j & 15), e = (kk & 7) + ((kk >> 4) << 3);
  int nks = Cpad >> 5;
  P[(((size_t)(jt * nks + ks)) * 32 + lane) * 16 + e] = (_Float16)v;
}

// ---------- f32 -> f16 ----------
__global__ void k_h2(const float* __restrict__ src, _Float16* __restrict__ dstp, int n) {
  int gid = blockIdx.x * blockDim.x + threadIdx.x;
  if (gid >= n) return;
  dstp[gid] = (_Float16)src[gid];
}

// ---------- WMMA GEMM: C[row, j] = act(A[row,:] . Bpack + bias) ----------
// One wave computes a 16x16 tile; K = KSTEPS*32; grids exact so EXEC is all-ones.
template <int KSTEPS, int LEAKY>
__global__ void k_gemm(const _Float16* __restrict__ A, const _Float16* __restrict__ Bp,
                       const float* __restrict__ bias, float* __restrict__ C,
                       int ldA, int ldC) {
  int wave = threadIdx.x >> 5;
  int lane = threadIdx.x & 31;
  int rowTile = blockIdx.x;
  int jTile = blockIdx.y * (blockDim.x >> 5) + wave;
  int hi = lane >> 4;
  int row = rowTile * 16 + (lane & 15);
  const _Float16* arow = A + (size_t)row * ldA + hi * 8;
  const _Float16* bp = Bp + (size_t)jTile * KSTEPS * 512 + (size_t)lane * 16;
  v8f acc = {};
#pragma unroll
  for (int ks = 0; ks < KSTEPS; ++ks) {
    // A fragment per ISA layout: lane holds row (lane&15), K = ks*32 + hi*8 + {0..7, 16..23}
    v8h alo = *(const v8h*)(arow + ks * 32);
    v8h ahi = *(const v8h*)(arow + ks * 32 + 16);
    v16h a = __builtin_shufflevector(alo, ahi, 0, 1, 2, 3, 4, 5, 6, 7,
                                     8, 9, 10, 11, 12, 13, 14, 15);
    v16h b = *(const v16h*)(bp + (size_t)ks * 512);
    acc = __builtin_amdgcn_wmma_f32_16x16x32_f16(false, a, false, b,
                                                 (short)0, acc, false, false);
  }
  int col = jTile * 16 + (lane & 15);
  float bj = bias[col];
#pragma unroll
  for (int r = 0; r < 8; ++r) {
    int orow = rowTile * 16 + hi * 8 + r;   // C/D layout: VGPR r -> row r + hi*8
    float v = acc[r] + bj;
    if (LEAKY) v = (v > 0.0f) ? v : 0.2f * v;
    C[(size_t)orow * ldC + col] = v;
  }
}

// ---------- BN partial sums: each block reduces 64 rows x 512 cols ----------
__global__ void k_bnstat(const float* __restrict__ h1, float* __restrict__ sum,
                         float* __restrict__ sq) {
  int j = threadIdx.x;          // 512 threads = one column each
  int r0 = blockIdx.x * 64;
  float s = 0.f, q = 0.f;
  for (int i = 0; i < 64; ++i) {
    int n = r0 + i;
    if (n >= NNODES) break;
    float v = h1[(size_t)n * HID + j];
    s += v; q += v * v;
  }
  unsafeAtomicAdd(&sum[j], s);
  unsafeAtomicAdd(&sq[j], q);
}

__global__ void k_bnfin(const float* __restrict__ sum, const float* __restrict__ sq,
                        const float* __restrict__ gamma, const float* __restrict__ beta,
                        float* __restrict__ scale, float* __restrict__ shift) {
  int j = threadIdx.x;
  float inv = 1.0f / (float)NNODES;
  float mu = sum[j] * inv;
  float var = sq[j] * inv - mu * mu;
  float rs = rsqrtf(var + 1e-5f);
  float sc = gamma[j] * rs;
  scale[j] = sc;
  shift[j] = beta[j] - mu * sc;
}

// ---------- apply BN + convert to f16 for GEMM2 ----------
__global__ void k_bnapply(const float* __restrict__ h1, const float* __restrict__ scale,
                          const float* __restrict__ shift, _Float16* __restrict__ Hn) {
  int gid = blockIdx.x * blockDim.x + threadIdx.x;
  if (gid >= NNODES * HID) return;
  int j = gid & (HID - 1);
  Hn[gid] = (_Float16)(h1[gid] * scale[j] + shift[j]);
}

// ---------- row L2-normalize z (one wave32 per node) ----------
__global__ void k_norm(const float* __restrict__ z, float* __restrict__ out) {
  int wid = (blockIdx.x * blockDim.x + threadIdx.x) >> 5;
  int lane = threadIdx.x & 31;
  if (wid >= NNODES) return;
  float a = z[(size_t)wid * 64 + lane];
  float b = z[(size_t)wid * 64 + 32 + lane];
  float s = a * a + b * b;
  for (int off = 16; off; off >>= 1) s += __shfl_xor(s, off, 32);
  float nrm = fmaxf(sqrtf(s), 1e-12f);
  out[(size_t)wid * 64 + lane] = a / nrm;
  out[(size_t)wid * 64 + 32 + lane] = b / nrm;
}

// ---------- r = z @ Wd.T + bd ----------
__global__ void k_r(const float* __restrict__ z, const float* __restrict__ Wd,
                    const float* __restrict__ bd, float* __restrict__ out) {
  int gid = blockIdx.x * blockDim.x + threadIdx.x;
  if (gid >= NNODES * 16) return;
  int n = gid >> 4, o = gid & 15;
  const float* zr = z + (size_t)n * 64;
  const float* wr = Wd + o * 64;
  float s = bd[o];
#pragma unroll
  for (int c = 0; c < 64; ++c) s += zr[c] * wr[c];
  out[gid] = s;
}

extern "C" void kernel_launch(void* const* d_in, const int* in_sizes, int n_in,
                              void* d_out, int out_size, void* d_ws, size_t ws_size,
                              hipStream_t stream) {
  (void)in_sizes; (void)n_in; (void)out_size; (void)ws_size;
  const float* feature = (const float*)d_in[0];
  const int*   ei      = (const int*)d_in[1];
  const float* ew      = (const float*)d_in[2];
  const int*   et      = (const int*)d_in[3];
  const float* W1      = (const float*)d_in[4];
  const float* b1      = (const float*)d_in[5];
  const float* gamma   = (const float*)d_in[6];
  const float* beta    = (const float*)d_in[7];
  const float* W2      = (const float*)d_in[8];
  const float* b2      = (const float*)d_in[9];
  const float* Wd      = (const float*)d_in[10];
  const float* bd      = (const float*)d_in[11];
  const int* src = ei;
  const int* dst = ei + NEDGES;

  char* ws = (char*)d_ws;
  float*    Hc  = (float*)(ws + HC_OFF);      // concat f32; reused as h1 after k_h2
  _Float16* Hh  = (_Float16*)(ws + HH_OFF);   // f16 concat; reused as BN-applied f16
  float*    z   = (float*)(ws + Z_OFF);
  _Float16* W1p = (_Float16*)(ws + W1P_OFF);
  _Float16* W2p = (_Float16*)(ws + W2P_OFF);
  float*    sums = (float*)(ws + SUM_OFF);
  float*    sqs  = (float*)(ws + SQ_OFF);
  float*    scl  = (float*)(ws + SC_OFF);
  float*    shf  = (float*)(ws + SH_OFF);

  // zero concat buffer (accumulated into) and BN stats; capturable memset nodes
  hipMemsetAsync(Hc, 0, (size_t)NNODES * CPAD * 4, stream);
  hipMemsetAsync(sums, 0, 4096, stream);  // sums + sqs contiguous

  k_feat<<<(NNODES * INDIM + 255) / 256, 256, 0, stream>>>(feature, Hc);
  k_packB<<<(HID * CPAD + 255) / 256, 256, 0, stream>>>(W1, W1p, HID, CONCAT, CPAD);
  k_packB<<<(OUTD * HID + 255) / 256, 256, 0, stream>>>(W2, W2p, OUTD, HID, HID);

  // K=4 propagation rounds into disjoint column slices of Hc
  const int bases[5] = {0, 16, 48, 112, 240};
  for (int k = 0; k < 4; ++k) {
    int Din = INDIM << k;
    int lg = 2 + k;                 // log2(Din/4)
    int total = NEDGES << lg;       // E * Din/4 threads
    k_scatter<<<(total + 255) / 256, 256, 0, stream>>>(src, dst, ew, et, Hc,
                                                       Din, bases[k], bases[k + 1], lg);
  }

  // f32 -> f16 for WMMA (padded columns stay zero)
  k_h2<<<(NNODES * CPAD + 255) / 256, 256, 0, stream>>>(Hc, Hh, NNODES * CPAD);

  // GEMM1 + bias + leaky_relu(0.2): h1 overwrites Hc region
  float* h1 = Hc;
  dim3 g1(NNODES / 16, 4);          // 3125 row tiles x (4 * 8 waves) = 32 j-tiles
  k_gemm<16, 1><<<g1, 256, 0, stream>>>(Hh, W1p, b1, h1, CPAD, HID);

  // BatchNorm (training-mode stats over N)
  k_bnstat<<<(NNODES + 63) / 64, 512, 0, stream>>>(h1, sums, sqs);
  k_bnfin<<<1, 512, 0, stream>>>(sums, sqs, gamma, beta, scl, shf);
  _Float16* Hn = Hh;                // BN-applied f16 overwrites Hh region
  k_bnapply<<<(NNODES * HID + 255) / 256, 256, 0, stream>>>(h1, scl, shf, Hn);

  // GEMM2 + bias -> z
  dim3 g2(NNODES / 16, 1);          // 4 j-tiles = 4 waves per block
  k_gemm<16, 0><<<g2, 128, 0, stream>>>(Hn, W2p, b2, z, HID, OUTD);

  // outputs: [z / ||z||  (N*64)] then [r (N*16)]
  float* out_z = (float*)d_out;
  float* out_r = out_z + (size_t)NNODES * OUTD;
  k_norm<<<NNODES / 8, 256, 0, stream>>>(z, out_z);
  k_r<<<(NNODES * 16 + 255) / 256, 256, 0, stream>>>(z, Wd, bd, out_r);
}
